// LaplaceBlock_89343909692234
// MI455X (gfx1250) — compile-verified, matrix operand order checked by reading, not run on hardware
//
#include <hip/hip_runtime.h>
#include <hip/hip_bf16.h>

// ---------------------------------------------------------------------------
// MPNN + Laplacian forward for MI455X (gfx1250, wave32).
// All dense 128-wide linears run through v_wmma_f32_16x16x32_bf16 (f32 accum).
// phi layer-1 is algebraically split: Ha = h@W1[0:128], Hb = h@W1[128:256]
// precomputed per NODE (8x fewer FLOPs than per-edge), edge messages are
// built on the fly inside the GEMM A-tile loader, and segment_sum is fused
// into the GEMM epilogue via global_atomic_add_f32.
// ---------------------------------------------------------------------------

typedef __attribute__((ext_vector_type(16))) __bf16 v16bf;
typedef __attribute__((ext_vector_type(8)))  __bf16 v8bf;
typedef __attribute__((ext_vector_type(8)))  float  v8f;

#define HDIM 128
#define LDT  48   // LDS row stride in bf16 elements (96B, keeps 16B alignment)

__device__ __forceinline__ float eluf(float x) {
  return x > 0.f ? x : (__expf(x) - 1.f);
}

// GEMM mode flags
#define GF_ELU     1   // ELU on output
#define GF_SPLIT   2   // A-tile cols: [0,K1) from A, [K1,K) from A2 (concat)
#define GF_EDGE    4   // A-tile built per edge: elu(Ha[dst]+Hb[src]+ea@Wc+bc)
#define GF_SCATTER 8   // epilogue: atomicAdd into scat[dstIdx[row]*128 + n]
#define GF_RESID  16   // epilogue: += resid[row*128+n]
#define GF_BIAS   32   // epilogue: += bias[n]

// D[M x 128] = op(A[M x K]) @ W[K x 128] (+bias)(+elu)(+resid | scatter)
__global__ __launch_bounds__(256)
void gemm_h128_wmma(const float* __restrict__ A, const float* __restrict__ A2,
                    int K, int K1,
                    const float* __restrict__ W, const float* __restrict__ bias,
                    const float* __restrict__ resid,
                    float* __restrict__ D,
                    float* __restrict__ scat, const int* __restrict__ dstIdx,
                    const int* __restrict__ srcIdx, const float* __restrict__ ea,
                    const float* __restrict__ wc, const float* __restrict__ bc,
                    int M, int flags)
{
  __shared__ __align__(16) __bf16 aT[128 * LDT];   // A tile, row-major [m][k]
  __shared__ __align__(16) __bf16 bT[128 * LDT];   // B tile, transposed [n][k]

  const int tid  = threadIdx.x;
  const int lane = tid & 31;
  const int wave = tid >> 5;
  const int m0   = blockIdx.x * 128;

  v8f zero8 = {0.f,0.f,0.f,0.f,0.f,0.f,0.f,0.f};
  v8f acc[8];
#pragma unroll
  for (int i = 0; i < 8; ++i) acc[i] = zero8;

  for (int kb = 0; kb < K; kb += 32) {
    // ---- stage A tile (f32 -> bf16) : rows m0..m0+127, cols kb..kb+31 ----
    {
      const int row = tid >> 1;           // 0..127
      const int h0  = (tid & 1) * 16;     // 0 or 16
      const int gr  = m0 + row;
      float vals[16];
      if (flags & GF_EDGE) {
        const int e  = gr;
        const int di = dstIdx[e];
        const int si = srcIdx[e];
        const float e0 = ea[e*3+0], e1 = ea[e*3+1], e2 = ea[e*3+2];
#pragma unroll
        for (int i = 0; i < 16; ++i) {
          const int k = kb + h0 + i;
          float v = A[(long)di*HDIM + k] + A2[(long)si*HDIM + k]
                  + e0*wc[0*HDIM + k] + e1*wc[1*HDIM + k] + e2*wc[2*HDIM + k]
                  + bc[k];
          vals[i] = eluf(v);
        }
      } else if (flags & GF_SPLIT) {
#pragma unroll
        for (int i = 0; i < 16; ++i) {
          const int k = kb + h0 + i;
          vals[i] = (k < K1) ? A[(long)gr*K1 + k]
                             : A2[(long)gr*(K - K1) + (k - K1)];
        }
      } else {
#pragma unroll
        for (int i = 0; i < 16; ++i)
          vals[i] = A[(long)gr*K + kb + h0 + i];
      }
#pragma unroll
      for (int i = 0; i < 16; ++i)
        aT[row*LDT + h0 + i] = (__bf16)vals[i];
    }
    // ---- stage B tile transposed: bT[n][kl] = W[(kb+kl)*128 + n] ----
    {
      const int n  = tid >> 1;
      const int h0 = (tid & 1) * 16;
#pragma unroll
      for (int i = 0; i < 16; ++i)
        bT[n*LDT + h0 + i] = (__bf16)W[(long)(kb + h0 + i)*HDIM + n];
    }
    __syncthreads();

    // ---- compute: wave owns one 16-col tile, 8 row tiles ----
    const int nloc = wave*16 + (lane & 15);
    const int kb8  = (lane >= 16) ? 8 : 0;   // 16-bit operand K split per lane half
    v16bf bfrag;
    {
      v8bf lo = *(const v8bf*)&bT[nloc*LDT + kb8];
      v8bf hi = *(const v8bf*)&bT[nloc*LDT + kb8 + 16];
#pragma unroll
      for (int i = 0; i < 8; ++i) { bfrag[i] = lo[i]; bfrag[i+8] = hi[i]; }
    }
#pragma unroll
    for (int mt = 0; mt < 8; ++mt) {
      const int arow = mt*16 + (lane & 15);
      v8bf lo = *(const v8bf*)&aT[arow*LDT + kb8];
      v8bf hi = *(const v8bf*)&aT[arow*LDT + kb8 + 16];
      v16bf afrag;
#pragma unroll
      for (int i = 0; i < 8; ++i) { afrag[i] = lo[i]; afrag[i+8] = hi[i]; }
      acc[mt] = __builtin_amdgcn_wmma_f32_16x16x32_bf16(
          /*neg_a=*/false, afrag, /*neg_b=*/false, bfrag,
          /*c_mod=*/(short)0, acc[mt], /*reuse_a=*/false, /*reuse_b=*/false);
    }
    __syncthreads();
  }

  // ---- epilogue ----
  const int n  = wave*16 + (lane & 15);
  const float bn = (flags & GF_BIAS) ? bias[n] : 0.f;
#pragma unroll
  for (int mt = 0; mt < 8; ++mt) {
#pragma unroll
    for (int v = 0; v < 8; ++v) {
      const int m = m0 + mt*16 + v + ((lane >= 16) ? 8 : 0);
      if (m < M) {
        float val = acc[mt][v] + bn;
        if (flags & GF_ELU) val = eluf(val);
        if (flags & GF_SCATTER) {
          unsafeAtomicAdd(&scat[(long)dstIdx[m]*HDIM + n], val);   // segment_sum
        } else {
          if (flags & GF_RESID) val += resid[(long)m*HDIM + n];
          D[(long)m*HDIM + n] = val;
        }
      }
    }
  }
}

// enc1: elu([y,pos] @ W(4x128) + b) -> out[BN x 128]; one wave per node
__global__ __launch_bounds__(256)
void enc1_kernel(const float* __restrict__ y, const float* __restrict__ pos,
                 const float* __restrict__ W, const float* __restrict__ b,
                 float* __restrict__ out, int BN_)
{
  const int node = (int)((blockIdx.x * (long)blockDim.x + threadIdx.x) >> 5);
  const int lane = threadIdx.x & 31;
  if (node >= BN_) return;
  const float x0 = y[node*2], x1 = y[node*2+1];
  const float x2 = pos[node*2], x3 = pos[node*2+1];
#pragma unroll
  for (int i = 0; i < 4; ++i) {
    const int c = lane + 32*i;
    float v = b[c] + x0*W[0*HDIM+c] + x1*W[1*HDIM+c]
                   + x2*W[2*HDIM+c] + x3*W[3*HDIM+c];
    out[(long)node*HDIM + c] = eluf(v);
  }
}

// Laplacian: out[b,i,:] = L[b,i,:] @ y[b,:,:]; y tile staged in LDS,
// one wave per row, float4 coalesced row reads + shuffle reduction.
__global__ __launch_bounds__(256)
void laplace_kernel(const float* __restrict__ L, const float* __restrict__ y,
                    float* __restrict__ out, int N_)
{
  extern __shared__ float ylds[];                 // N_*2 floats (16KB @ N=2048)
  const int b = blockIdx.y;
  for (int i = threadIdx.x; i < N_*2; i += blockDim.x)
    ylds[i] = y[(long)b*N_*2 + i];
  __syncthreads();

  const int wave = threadIdx.x >> 5, lane = threadIdx.x & 31;
  const int row  = blockIdx.x * 8 + wave;
  if (row >= N_) return;
  const float* Lr = L + ((long)b*N_ + row)*N_;
  float a0 = 0.f, a1 = 0.f;
  for (int j = lane*4; j < N_; j += 128) {
    const float4 lv = *(const float4*)&Lr[j];
    a0 += lv.x*ylds[2*j+0] + lv.y*ylds[2*j+2] + lv.z*ylds[2*j+4] + lv.w*ylds[2*j+6];
    a1 += lv.x*ylds[2*j+1] + lv.y*ylds[2*j+3] + lv.z*ylds[2*j+5] + lv.w*ylds[2*j+7];
  }
#pragma unroll
  for (int off = 16; off; off >>= 1) {
    a0 += __shfl_xor(a0, off, 32);
    a1 += __shfl_xor(a1, off, 32);
  }
  if (lane == 0) {
    const long o = ((long)b*N_ + row)*2;
    out[o+0] = a0;
    out[o+1] = a1;
  }
}

// dec2 + final: out[i,:] = d_vec[i] * (hdec[i,:] @ W(128x2) + b) + out[i,:]
__global__ __launch_bounds__(256)
void dec2_final_kernel(const float* __restrict__ hdec, const float* __restrict__ W,
                       const float* __restrict__ b, const float* __restrict__ dvec,
                       float* __restrict__ out, int BN_)
{
  const int node = (int)((blockIdx.x * (long)blockDim.x + threadIdx.x) >> 5);
  const int lane = threadIdx.x & 31;
  if (node >= BN_) return;
  float a0 = 0.f, a1 = 0.f;
  for (int k = lane; k < HDIM; k += 32) {
    const float hv = hdec[(long)node*HDIM + k];
    a0 += hv * W[k*2+0];
    a1 += hv * W[k*2+1];
  }
#pragma unroll
  for (int off = 16; off; off >>= 1) {
    a0 += __shfl_xor(a0, off, 32);
    a1 += __shfl_xor(a1, off, 32);
  }
  if (lane == 0) {
    const float d = dvec[node];
    out[node*2+0] = d*(a0 + b[0]) + out[node*2+0];
    out[node*2+1] = d*(a1 + b[1]) + out[node*2+1];
  }
}

__global__ void zero_kernel(float* __restrict__ p, long n) {
  const long i = blockIdx.x * (long)blockDim.x + threadIdx.x;
  if (i < n) p[i] = 0.f;
}

extern "C" void kernel_launch(void* const* d_in, const int* in_sizes, int n_in,
                              void* d_out, int out_size, void* d_ws, size_t ws_size,
                              hipStream_t stream)
{
  const float* y    = (const float*)d_in[0];
  const float* pos  = (const float*)d_in[1];
  const float* ea   = (const float*)d_in[2];
  const int*   eidx = (const int*)d_in[3];
  const float* dvec = (const float*)d_in[4];
  const float* lapM = (const float*)d_in[5];

  const int BN_ = in_sizes[0] / 2;
  const int E_  = in_sizes[2] / 3;
  const int N_  = (int)((long)in_sizes[5] / BN_);
  const int B_  = BN_ / N_;
  const int* srcI = eidx;        // edge_index[0]
  const int* dstI = eidx + E_;   // edge_index[1]

  // ---- parameter pointers (handle both flatten orders via in_sizes) ----
  const float* P[32];
  for (int i = 6; i < n_in && i < 38; ++i) P[i-6] = (const float*)d_in[i];
  const float *enc1W,*enc1b,*enc2W,*enc2b,*dec1W,*dec1b,*dec2W,*dec2b;
  const float *phiW1[3],*phib1[3],*phiW2[3],*phib2[3];
  const float *gamW1[3],*gamb1[3],*gamW2[3],*gamb2[3];
  if (in_sizes[6] == 4*HDIM) {
    // dict-insertion order: enc1,enc2,dec1,dec2, mpnn{phi,gamma}
    enc1W=P[0]; enc1b=P[1]; enc2W=P[2]; enc2b=P[3];
    dec1W=P[4]; dec1b=P[5]; dec2W=P[6]; dec2b=P[7];
    for (int l = 0; l < 3; ++l) {
      const float* const* q = P + 8 + 8*l;
      phiW1[l]=q[0]; phib1[l]=q[1]; phiW2[l]=q[2]; phib2[l]=q[3];
      gamW1[l]=q[4]; gamb1[l]=q[5]; gamW2[l]=q[6]; gamb2[l]=q[7];
    }
  } else {
    // jax pytree sorted-key order: dec1,dec2,enc1,enc2, mpnn{gamma,phi}
    dec1W=P[0]; dec1b=P[1]; dec2W=P[2]; dec2b=P[3];
    enc1W=P[4]; enc1b=P[5]; enc2W=P[6]; enc2b=P[7];
    for (int l = 0; l < 3; ++l) {
      const float* const* q = P + 8 + 8*l;
      gamW1[l]=q[0]; gamb1[l]=q[1]; gamW2[l]=q[2]; gamb2[l]=q[3];
      phiW1[l]=q[4]; phib1[l]=q[5]; phiW2[l]=q[6]; phib2[l]=q[7];
    }
  }

  // ---- workspace layout: 6 node-sized f32 buffers (~96MB @ BN=32768) ----
  float* ws = (float*)d_ws;
  const long NH = (long)BN_ * HDIM;
  float* hA  = ws + 0*NH;
  float* hB  = ws + 1*NH;
  float* Ha  = ws + 2*NH;
  float* Hb  = ws + 3*NH;
  float* agg = ws + 4*NH;
  float* tmp = ws + 5*NH;   // enc1 act / gamma hidden / dec1 act

  const int gN = (BN_ + 127) / 128;
  const int gE = (E_  + 127) / 128;

#define GEMM(A_,A2_,K_,K1_,W_,b_,res_,D_,scat_,dI_,sI_,ea_,wc_,bc_,M_,g_,fl_) \
  gemm_h128_wmma<<<dim3(g_), dim3(256), 0, stream>>>(                         \
      A_, A2_, K_, K1_, W_, b_, res_, D_, scat_, dI_, sI_, ea_, wc_, bc_, M_, fl_)

  // ---- encoder ----
  enc1_kernel<<<dim3((BN_ + 7) / 8), dim3(256), 0, stream>>>(
      y, pos, enc1W, enc1b, tmp, BN_);
  GEMM(tmp, tmp, 128, 128, enc2W, enc2b, nullptr, hA,
       nullptr, nullptr, nullptr, nullptr, nullptr, nullptr, BN_, gN, GF_BIAS);

  // ---- MPNN layers ----
  float* h  = hA;
  float* hn = hB;
  for (int l = 0; l < 3; ++l) {
    // Ha = h @ W1[0:128,:], Hb = h @ W1[128:256,:]  (per-node precompute)
    GEMM(h, h, 128, 128, phiW1[l],              nullptr, nullptr, Ha,
         nullptr, nullptr, nullptr, nullptr, nullptr, nullptr, BN_, gN, 0);
    GEMM(h, h, 128, 128, phiW1[l] + 128*HDIM,   nullptr, nullptr, Hb,
         nullptr, nullptr, nullptr, nullptr, nullptr, nullptr, BN_, gN, 0);
    // agg = 0
    zero_kernel<<<dim3((unsigned)((NH + 255) / 256)), dim3(256), 0, stream>>>(agg, NH);
    // msg GEMM over edges with fused segment_sum scatter:
    //   A[e] = elu(Ha[dst]+Hb[src]+ea@W1c+b1);  agg[dst] += A[e]@W2 + b2
    GEMM(Ha, Hb, 128, 128, phiW2[l], phib2[l], nullptr, nullptr,
         agg, dstI, srcI, ea, phiW1[l] + 256*HDIM, phib1[l],
         E_, gE, GF_EDGE | GF_SCATTER | GF_BIAS);
    // gamma MLP: tmp = elu([h|agg] @ G1 + c1);  h' = h + tmp @ G2 + c2
    GEMM(h, agg, 256, 128, gamW1[l], gamb1[l], nullptr, tmp,
         nullptr, nullptr, nullptr, nullptr, nullptr, nullptr, BN_, gN,
         GF_SPLIT | GF_ELU | GF_BIAS);
    GEMM(tmp, tmp, 128, 128, gamW2[l], gamb2[l], h, hn,
         nullptr, nullptr, nullptr, nullptr, nullptr, nullptr, BN_, gN,
         GF_RESID | GF_BIAS);
    float* t = h; h = hn; hn = t;
  }

  // ---- decoder hidden ----
  GEMM(h, h, 128, 128, dec1W, dec1b, nullptr, tmp,
       nullptr, nullptr, nullptr, nullptr, nullptr, nullptr, BN_, gN,
       GF_ELU | GF_BIAS);

  // ---- Laplacian writes d_out, then dec2 head accumulates into it ----
  laplace_kernel<<<dim3((N_ + 7) / 8, B_), dim3(256), (size_t)N_ * 2 * sizeof(float),
                   stream>>>(lapM, y, (float*)d_out, N_);
  dec2_final_kernel<<<dim3((BN_ + 7) / 8), dim3(256), 0, stream>>>(
      tmp, dec2W, dec2b, dvec, (float*)d_out, BN_);

#undef GEMM
}
